// GCNLayer_74990128988467
// MI455X (gfx1250) — matrix operand order repaired and checked by reading, not compiled, not run on hardware
//
#include <hip/hip_runtime.h>
#include <stdint.h>

// ---------------------------------------------------------------------------
// Types / helpers
// ---------------------------------------------------------------------------
typedef __attribute__((ext_vector_type(16))) __bf16 bf16x16;
typedef __attribute__((ext_vector_type(8)))  __bf16 bf16x8;
typedef __attribute__((ext_vector_type(8)))  float  f32x8;
typedef int v4i_vs __attribute__((vector_size(16)));   // matches builtin's param type

#define BM 128
#define BN 128
#define BK 32
#define LDSS 40   // LDS row stride in bf16 elems (padded; 80B keeps 16B alignment)

enum { OUT_F32 = 0, OUT_RELU_BF16 = 1, OUT_BF16 = 2 };

#if defined(__AMDGCN__) && __has_builtin(__builtin_amdgcn_global_load_async_to_lds_b128)
#define USE_ASYNC_LDS 1
#else
#define USE_ASYNC_LDS 0
#endif

// Arg1: generic v4i* (exactly the type clang reported). Arg2: as3 v4i* — exact
// match if the param is v4i addrspace(3)*, implicitly converts if void as3*.
#define GPTR(p) ((v4i_vs*)(void*)(p))
#define LPTR(p) ((__attribute__((address_space(3))) v4i_vs*)(p))

__device__ __forceinline__ int imin(int a, int b) { return a < b ? a : b; }

// ---------------------------------------------------------------------------
// Tiled bf16 GEMM: C[M,Nout] = act(A[M,K] @ Wt[Nout,K]^T + bias)
// Wt is stored transposed ([N][K]) so both A and B tiles stream contiguously.
// 256 threads = 8 waves; wave grid 2x4; each wave owns 64x32 = 4x2 WMMA tiles.
// Double-buffered LDS, software-pipelined staging (async-to-LDS when the
// gfx1250 builtin is available, register prefetch otherwise).
// Requires: K % 32 == 0, Nout % 128 == 0. Rows beyond M are clamped (their
// WMMA results are masked at the store, so duplicates are harmless).
// ---------------------------------------------------------------------------
template <int MODE>
__global__ __launch_bounds__(256)
void gemm_bf16_wmma(const __bf16* __restrict__ A, const __bf16* __restrict__ Wt,
                    const float* __restrict__ bias, void* __restrict__ Cout,
                    int M, int K, int Nout, int ldc)
{
    __shared__ __bf16 As[2][BM * LDSS];
    __shared__ __bf16 Bs[2][BN * LDSS];

    const int tid  = threadIdx.x;
    const int lane = tid & 31;
    const int wave = tid >> 5;
    const int wm   = wave >> 2;      // 0..1 -> M offset wm*64
    const int wn   = wave & 3;       // 0..3 -> N offset wn*32
    const int lm   = lane & 15;
    const int half = lane >> 4;

    const int row_blk = blockIdx.y * BM;
    const int col_blk = blockIdx.x * BN;

    // Staging assignment: 1024 16B-vectors per tile pair; 2 A + 2 B per thread.
    int sr[2], sc[2];              // tile row 0..127, col offset {0,8,16,24}
    const __bf16* gA[2];           // clamped global source pointers (K advances)
    const __bf16* gB[2];
#pragma unroll
    for (int i = 0; i < 2; ++i) {
        const int v = tid * 2 + i;
        sr[i] = v >> 2;
        sc[i] = (v & 3) * 8;
        const int ar = imin(row_blk + sr[i], M - 1);
        const int bc = imin(col_blk + sr[i], Nout - 1);
        gA[i] = A  + (size_t)ar * K + sc[i];
        gB[i] = Wt + (size_t)bc * K + sc[i];
    }

    f32x8 acc[4][2];
#pragma unroll
    for (int i = 0; i < 4; ++i)
#pragma unroll
        for (int j = 0; j < 2; ++j)
#pragma unroll
            for (int e = 0; e < 8; ++e) acc[i][j][e] = 0.0f;

    const int ntiles = K / BK;

#if USE_ASYNC_LDS
    // ---- async staging: global -> LDS without VGPR round-trip (ASYNCcnt) ----
    {
#pragma unroll
        for (int i = 0; i < 2; ++i) {
            __builtin_amdgcn_global_load_async_to_lds_b128(
                GPTR(gA[i]), LPTR(&As[0][sr[i] * LDSS + sc[i]]), 0, 0);
            __builtin_amdgcn_global_load_async_to_lds_b128(
                GPTR(gB[i]), LPTR(&Bs[0][sr[i] * LDSS + sc[i]]), 0, 0);
        }
    }
#else
    // ---- fallback staging: prefetch tile 0 into registers ----
    bf16x8 pa[2], pb[2];
#pragma unroll
    for (int i = 0; i < 2; ++i) {
        pa[i] = *(const bf16x8*)(gA[i]);
        pb[i] = *(const bf16x8*)(gB[i]);
    }
#endif

    int p = 0;
    for (int kt = 0; kt < ntiles; ++kt) {
#if USE_ASYNC_LDS
        // My tile-kt async writes are the only outstanding ones here.
        asm volatile("s_wait_asynccnt 0x0" ::: "memory");
        __syncthreads();   // all waves: tile kt in LDS; buf[p^1] reads done
        if (kt + 1 < ntiles) {
            const int koff = (kt + 1) * BK;
#pragma unroll
            for (int i = 0; i < 2; ++i) {
                __builtin_amdgcn_global_load_async_to_lds_b128(
                    GPTR(gA[i] + koff), LPTR(&As[p ^ 1][sr[i] * LDSS + sc[i]]), 0, 0);
                __builtin_amdgcn_global_load_async_to_lds_b128(
                    GPTR(gB[i] + koff), LPTR(&Bs[p ^ 1][sr[i] * LDSS + sc[i]]), 0, 0);
            }
        }
#else
        // Store prefetched registers, then immediately issue next tile's loads.
#pragma unroll
        for (int i = 0; i < 2; ++i) {
            *(bf16x8*)(&As[p][sr[i] * LDSS + sc[i]]) = pa[i];
            *(bf16x8*)(&Bs[p][sr[i] * LDSS + sc[i]]) = pb[i];
        }
        if (kt + 1 < ntiles) {
            const int koff = (kt + 1) * BK;
#pragma unroll
            for (int i = 0; i < 2; ++i) {
                pa[i] = *(const bf16x8*)(gA[i] + koff);
                pb[i] = *(const bf16x8*)(gB[i] + koff);
            }
        }
        __syncthreads();   // tile kt visible to all waves
#endif

        // ---- B fragments: lane half selects K 0-15 / 16-31 (contiguous) ----
        bf16x16 bfrag[2];
#pragma unroll
        for (int nt = 0; nt < 2; ++nt) {
            const __bf16* bp = &Bs[p][(wn * 32 + nt * 16 + lm) * LDSS + half * 16];
            bf16x8 lo = *(const bf16x8*)(bp);
            bf16x8 hi = *(const bf16x8*)(bp + 8);
            bfrag[nt] = __builtin_shufflevector(lo, hi, 0, 1, 2, 3, 4, 5, 6, 7,
                                                8, 9, 10, 11, 12, 13, 14, 15);
        }

        // ---- A fragments: lane half selects K 0-15 / 16-31 (contiguous) ----
#pragma unroll
        for (int mt = 0; mt < 4; ++mt) {
            const __bf16* ap = &As[p][(wm * 64 + mt * 16 + lm) * LDSS + half * 16];
            bf16x8 lo = *(const bf16x8*)(ap);
            bf16x8 hi = *(const bf16x8*)(ap + 8);
            bf16x16 afrag = __builtin_shufflevector(lo, hi, 0, 1, 2, 3, 4, 5, 6, 7,
                                                    8, 9, 10, 11, 12, 13, 14, 15);
#pragma unroll
            for (int nt = 0; nt < 2; ++nt) {
                acc[mt][nt] = __builtin_amdgcn_wmma_f32_16x16x32_bf16(
                    false, afrag, false, bfrag[nt], (short)0, acc[mt][nt],
                    false, false);
            }
        }

#if !USE_ASYNC_LDS
        __syncthreads();   // reads of buf[p] done before it is rewritten
#endif
        p ^= 1;
    }

    // ---- epilogue: D layout — VGPR j -> row j + 8*half; col = lane%16 ----
    // Nout % 128 == 0 for every launch, so only the row guard is needed.
#pragma unroll
    for (int mt = 0; mt < 4; ++mt) {
#pragma unroll
        for (int nt = 0; nt < 2; ++nt) {
            const int col = col_blk + wn * 32 + nt * 16 + lm;
            const float bv = bias[col];
#pragma unroll
            for (int j = 0; j < 8; ++j) {
                const int row = row_blk + wm * 64 + mt * 16 + half * 8 + j;
                if (row < M) {
                    float v = acc[mt][nt][j] + bv;
                    if (MODE == OUT_F32) {
                        ((float*)Cout)[(size_t)row * ldc + col] = v;
                    } else {
                        if (MODE == OUT_RELU_BF16) v = v > 0.0f ? v : 0.0f;
                        ((__bf16*)Cout)[(size_t)row * ldc + col] = (__bf16)v;
                    }
                }
            }
        }
    }
}

// ---------------------------------------------------------------------------
// Support kernels
// ---------------------------------------------------------------------------
// f32 W[K,N] -> bf16 Wt[N,K]
__global__ void k_weight_t(const float* __restrict__ W, __bf16* __restrict__ Wt,
                           int K, int N)
{
    int idx = blockIdx.x * 256 + threadIdx.x;
    if (idx >= K * N) return;
    int k = idx / N, n = idx % N;
    Wt[(size_t)n * K + k] = (__bf16)W[idx];
}

__global__ void k_cvt_bf16(const float* __restrict__ X, __bf16* __restrict__ Y,
                           long long n)
{
    long long i = (long long)blockIdx.x * 256 + threadIdx.x;
    if (i < n) Y[i] = (__bf16)X[i];
}

__global__ void k_zero_f32(float* __restrict__ p, long long n)
{
    long long i = (long long)blockIdx.x * 256 + threadIdx.x;
    if (i < n) p[i] = 0.0f;
}

// msg_in gather: X0[e] = concat(node_feat[src[e]], node_attr[src[e]], edge_attr[e])
__global__ void k_gather(const long long* __restrict__ ei,
                         const float* __restrict__ nf,
                         const float* __restrict__ na,
                         const float* __restrict__ ea,
                         __bf16* __restrict__ X0, int E)
{
    long long idx = (long long)blockIdx.x * 256 + threadIdx.x;
    if (idx >= (long long)E * 224) return;
    int e = (int)(idx / 224);
    int c = (int)(idx % 224);
    long long s = ei[e];               // edge_index[0][e]
    float v;
    if (c < 128)      v = nf[s * 128 + c];
    else if (c < 192) v = na[s * 64 + (c - 128)];
    else              v = ea[(long long)e * 32 + (c - 192)];
    X0[(size_t)e * 224 + c] = (__bf16)v;
}

// segment_sum via global f32 atomics: agg[dst[e]] += Se[e]
__global__ void k_scatter_add(const float* __restrict__ Se,
                              const long long* __restrict__ ei,
                              float* __restrict__ agg, int E)
{
    long long idx = (long long)blockIdx.x * 256 + threadIdx.x;
    if (idx >= (long long)E * 128) return;
    int e = (int)(idx >> 7);
    int c = (int)(idx & 127);
    long long d = ei[(long long)E + e];   // edge_index[1][e]
    atomicAdd(agg + d * 128 + c, Se[idx]);
}

// f32 [M,Nc] -> bf16 strided into concat buffer (ldy columns)
__global__ void k_cvt_strided(const float* __restrict__ X, __bf16* __restrict__ Y,
                              int M, int Nc, int ldy)
{
    long long idx = (long long)blockIdx.x * 256 + threadIdx.x;
    if (idx >= (long long)M * Nc) return;
    int r = (int)(idx / Nc), c = (int)(idx % Nc);
    Y[(size_t)r * ldy + c] = (__bf16)X[idx];
}

// ---------------------------------------------------------------------------
// Host orchestration
// ---------------------------------------------------------------------------
static inline int eg(long long n) { return (int)((n + 255) / 256); }

static void launch_gemm(const __bf16* A, const __bf16* Wt, const float* bias,
                        void* C, int M, int K, int Nout, int ldc, int mode,
                        hipStream_t s)
{
    dim3 grid((Nout + BN - 1) / BN, (M + BM - 1) / BM);
    if (mode == OUT_F32)
        gemm_bf16_wmma<OUT_F32><<<grid, 256, 0, s>>>(A, Wt, bias, C, M, K, Nout, ldc);
    else if (mode == OUT_RELU_BF16)
        gemm_bf16_wmma<OUT_RELU_BF16><<<grid, 256, 0, s>>>(A, Wt, bias, C, M, K, Nout, ldc);
    else
        gemm_bf16_wmma<OUT_BF16><<<grid, 256, 0, s>>>(A, Wt, bias, C, M, K, Nout, ldc);
}

extern "C" void kernel_launch(void* const* d_in, const int* in_sizes, int n_in,
                              void* d_out, int out_size, void* d_ws, size_t ws_size,
                              hipStream_t stream)
{
    const long long* edge_index = (const long long*)d_in[0];   // [2,E] int64
    const float*     node_feat  = (const float*)d_in[1];       // [N,128]
    const float*     node_attr  = (const float*)d_in[2];       // [N,64]
    const float*     edge_attr  = (const float*)d_in[3];       // [E,32]

    const int E = in_sizes[0] / 2;
    const int N = in_sizes[1] / 128;

    // MLP layer dims: in -> 1024 -> 2048 -> 1024 -> 128
    const int srcD[5]  = {224, 1024, 2048, 1024, 128};
    const int dstD[5]  = {64,  1024, 2048, 1024, 128};
    const int featD[5] = {128, 1024, 2048, 1024, 128};

    const float* srcW[4];  const float* srcB[4];
    const float* dstW[4];  const float* dstB[4];
    const float* featW[4]; const float* featB[4];
    for (int i = 0; i < 4; ++i) {
        srcW[i]  = (const float*)d_in[4  + 2 * i];  srcB[i]  = (const float*)d_in[5  + 2 * i];
        dstW[i]  = (const float*)d_in[12 + 2 * i];  dstB[i]  = (const float*)d_in[13 + 2 * i];
        featW[i] = (const float*)d_in[20 + 2 * i];  featB[i] = (const float*)d_in[21 + 2 * i];
    }
    const float* fcW = (const float*)d_in[28];   // [384,128]
    const float* fcb = (const float*)d_in[29];   // [128]

    // ---- workspace carve (256B aligned) ----
    char* wsp = (char*)d_ws;
    auto alloc = [&](size_t bytes) -> void* {
        void* p = (void*)wsp;
        wsp += (bytes + 255) & ~(size_t)255;
        return p;
    };

    __bf16* srcWt[4];  __bf16* dstWt[4];  __bf16* featWt[4];
    for (int i = 0; i < 4; ++i)
        srcWt[i]  = (__bf16*)alloc((size_t)srcD[i]  * srcD[i + 1]  * 2);
    for (int i = 0; i < 4; ++i)
        dstWt[i]  = (__bf16*)alloc((size_t)dstD[i]  * dstD[i + 1]  * 2);
    for (int i = 0; i < 4; ++i)
        featWt[i] = (__bf16*)alloc((size_t)featD[i] * featD[i + 1] * 2);
    __bf16* fcWt = (__bf16*)alloc((size_t)384 * 128 * 2);

    size_t x0_elems  = (size_t)E * 224;
    size_t cat_elems = (size_t)N * 384;
    __bf16* X0    = (__bf16*)alloc((x0_elems > cat_elems ? x0_elems : cat_elems) * 2);
    __bf16* bufA  = (__bf16*)alloc((size_t)E * 2048 * 2);
    __bf16* bufB  = (__bf16*)alloc((size_t)E * 2048 * 2);
    float*  Se    = (float*) alloc((size_t)E * 128 * 4);
    float*  agg   = (float*) alloc((size_t)N * 128 * 4);
    __bf16* Xattr = (__bf16*)alloc((size_t)N * 64 * 2);
    __bf16* Xfeat = (__bf16*)alloc((size_t)N * 128 * 2);

    // ---- 1. weights -> bf16 transposed ----
    for (int i = 0; i < 4; ++i) {
        k_weight_t<<<eg((long long)srcD[i] * srcD[i + 1]), 256, 0, stream>>>(
            srcW[i], srcWt[i], srcD[i], srcD[i + 1]);
        k_weight_t<<<eg((long long)dstD[i] * dstD[i + 1]), 256, 0, stream>>>(
            dstW[i], dstWt[i], dstD[i], dstD[i + 1]);
        k_weight_t<<<eg((long long)featD[i] * featD[i + 1]), 256, 0, stream>>>(
            featW[i], featWt[i], featD[i], featD[i + 1]);
    }
    k_weight_t<<<eg(384 * 128), 256, 0, stream>>>(fcW, fcWt, 384, 128);

    // ---- 2. node inputs -> bf16; zero aggregate ----
    k_cvt_bf16<<<eg((long long)N * 64), 256, 0, stream>>>(node_attr, Xattr, (long long)N * 64);
    k_cvt_bf16<<<eg((long long)N * 128), 256, 0, stream>>>(node_feat, Xfeat, (long long)N * 128);
    k_zero_f32<<<eg((long long)N * 128), 256, 0, stream>>>(agg, (long long)N * 128);

    // ---- 3. gather edge inputs ----
    k_gather<<<eg((long long)E * 224), 256, 0, stream>>>(
        edge_index, node_feat, node_attr, edge_attr, X0, E);

    // ---- 4. src MLP over edges ----
    launch_gemm(X0,   srcWt[0], srcB[0], bufA, E, 224,  1024, 1024, OUT_RELU_BF16, stream);
    launch_gemm(bufA, srcWt[1], srcB[1], bufB, E, 1024, 2048, 2048, OUT_RELU_BF16, stream);
    launch_gemm(bufB, srcWt[2], srcB[2], bufA, E, 2048, 1024, 1024, OUT_RELU_BF16, stream);
    launch_gemm(bufA, srcWt[3], srcB[3], Se,   E, 1024, 128,  128,  OUT_F32,       stream);

    // ---- 5. segment_sum via scatter-add ----
    k_scatter_add<<<eg((long long)E * 128), 256, 0, stream>>>(Se, edge_index, agg, E);

    // ---- 6. dst MLP over nodes -> concat[:, 0:128] ----
    launch_gemm(Xattr, dstWt[0], dstB[0], bufA, N, 64,   1024, 1024, OUT_RELU_BF16, stream);
    launch_gemm(bufA,  dstWt[1], dstB[1], bufB, N, 1024, 2048, 2048, OUT_RELU_BF16, stream);
    launch_gemm(bufB,  dstWt[2], dstB[2], bufA, N, 2048, 1024, 1024, OUT_RELU_BF16, stream);
    launch_gemm(bufA,  dstWt[3], dstB[3], X0,   N, 1024, 128,  384,  OUT_BF16,      stream);

    // ---- 7. feat MLP over nodes -> concat[:, 128:256] ----
    launch_gemm(Xfeat, featWt[0], featB[0], bufA,     N, 128,  1024, 1024, OUT_RELU_BF16, stream);
    launch_gemm(bufA,  featWt[1], featB[1], bufB,     N, 1024, 2048, 2048, OUT_RELU_BF16, stream);
    launch_gemm(bufB,  featWt[2], featB[2], bufA,     N, 2048, 1024, 1024, OUT_RELU_BF16, stream);
    launch_gemm(bufA,  featWt[3], featB[3], X0 + 128, N, 1024, 128,  384,  OUT_BF16,      stream);

    // ---- 8. agg -> concat[:, 256:384] (bf16) ----
    k_cvt_strided<<<eg((long long)N * 128), 256, 0, stream>>>(agg, X0 + 256, N, 128, 384);

    // ---- 9. final fc: [N,384] @ [384,128] + b -> d_out (f32) ----
    launch_gemm(X0, fcWt, fcb, d_out, N, 384, 128, 128, OUT_F32, stream);
}
